// GVPModel_44014824849933
// MI455X (gfx1250) — matrix operand order repaired and checked by reading, not compiled, not run on hardware
//
#include <hip/hip_runtime.h>
#include <hip/hip_bf16.h>
#include <math.h>

// ---------------------------------------------------------------------------
// GVP graph network on MI455X (gfx1250, wave32).
// Heavy GEMMs (edge msg0/msg1, node ff0/ff1) run on v_wmma_f32_16x16x32_f16
// with f16 inputs / f32 accum; everything else (bessel features, layernorms,
// sigmoid gates, small 33x33 vector mixes, atomic scatter) is f32 VALU.
// Weight matrices are pre-swizzled to the WMMA B-fragment layout each launch.
// params flattening assumed = jax tree_leaves (sorted dict keys, lists in
// order) after the 5 array inputs.
// ---------------------------------------------------------------------------

typedef __attribute__((ext_vector_type(16))) _Float16 v16h;
typedef __attribute__((ext_vector_type(8)))  _Float16 v8h;
typedef __attribute__((ext_vector_type(8)))  float    v8f;

#define NNODE  30000
#define NEDGE  480000
#define NSC    128
#define NVC    16
#define NGRAPH 16
#define NOUT   64
#define EPSF   1e-8f

__device__ __forceinline__ float sigmoidf_(float x){ return 1.f/(1.f+__expf(-x)); }
__device__ __forceinline__ float siluf_(float x){ return x*sigmoidf_(x); }

// ---------------------------------------------------------------------------
__global__ __launch_bounds__(256) void k_zero(float* p, int n){
  int i = blockIdx.x*256 + threadIdx.x;
  if(i < n) p[i] = 0.f;
}

// f32 KxN row-major -> f16 WMMA-B swizzled tiles.
// dst[((kt*NT+nt)*32 + lane)*16 + h] = src[k*N + n],
//   n = nt*16 + lane%16, k = kt*32 + 8*(lane/16) + h%8 + 16*(h/8)  (zero-pad k>=K)
__global__ __launch_bounds__(256) void k_swizzle(const float* __restrict__ src,
                                                 _Float16* __restrict__ dst,
                                                 int K, int N, int KT){
  int idx = blockIdx.x*256 + threadIdx.x;
  int total = KT*(N>>4)*512;
  if(idx >= total) return;
  int h    = idx & 15;
  int lane = (idx>>4) & 31;
  int tile = idx >> 9;
  int NT = N >> 4;
  int kt = tile / NT, nt = tile % NT;
  int n = nt*16 + (lane & 15);
  int k = kt*32 + ((lane>>4)<<3) + (h&7) + ((h>>3)<<4);
  dst[idx] = (_Float16)((k < K) ? src[k*N + n] : 0.f);
}

// ---------------------------------------------------------------------------
// One 16x16 C tile: A is LDS row-major f16 [16][KP], B is swizzled in global.
// A fragment per ISA layout: lane L holds row m=L%16; halves 0..7 -> k=kb..kb+7,
// halves 8..15 -> k=kb+16..kb+23, kb = 8*(L/16).
template<int KT, int NT, int KP>
__device__ __forceinline__ v8f wmma_acc(const _Float16* Ash, int m,
                                        const _Float16* __restrict__ Bsw, int nt){
  const int lane = threadIdx.x & 31;
  const int kb   = (lane >> 4) * 8;
  const _Float16* row = Ash + m*KP + kb;
  v8f c = {};
  #pragma unroll
  for(int kt = 0; kt < KT; ++kt){
    v8h lo = *(const v8h*)(row + kt*32);
    v8h hi = *(const v8h*)(row + kt*32 + 16);
    union { v16h v; v8h p[2]; } au;
    au.p[0] = lo; au.p[1] = hi;
    v16h b = *(const v16h*)(Bsw + ((size_t)(kt*NT + nt)*32 + lane)*16);
    c = __builtin_amdgcn_wmma_f32_16x16x32_f16(false, au.v, false, b,
                                               (short)0, c, false, false);
  }
  return c;
}

// ---------------------------------------------------------------------------
struct EdgeFeatP {
  const float* pos; const float* shifts;
  const int* src; const int* dst;
  const float* lng; const float* lnb;           // We_ln
  const float* wh; const float* ws_w; const float* ws_b;   // We
  const float* wv; const float* wsv_w; const float* wsv_b;
  float* es; float* ev; float* deg;
};

__global__ __launch_bounds__(256) void k_edge_feat(EdgeFeatP P){
  int e = blockIdx.x*256 + threadIdx.x;
  if(e >= NEDGE) return;
  int s = P.src[e], d = P.dst[e];
  float vx = P.pos[d*3+0] - P.pos[s*3+0] + P.shifts[e*3+0];
  float vy = P.pos[d*3+1] - P.pos[s*3+1] + P.shifts[e*3+1];
  float vz = P.pos[d*3+2] - P.pos[s*3+2] + P.shifts[e*3+2];
  float l2 = fmaxf(vx*vx + vy*vy + vz*vz, EPSF);
  float L  = sqrtf(l2);
  float ux = vx/L, uy = vy/L, uz = vz/L;
  const float PIf = 3.14159265358979f;
  float x  = L * 0.2f;                  // L / CUTOFF
  float x2 = x*x, x3 = x2*x, x6 = x3*x3, x7 = x6*x, x8 = x7*x;
  float env = (x < 1.f) ? (1.f - 28.f*x6 + 48.f*x7 - 21.f*x8) : 0.f;
  float sb = sqrtf(0.4f);
  float es0[8];
  float mu = 0.f;
  #pragma unroll
  for(int i = 0; i < 8; ++i){
    es0[i] = sb * __sinf((float)(i+1)*PIf*x) / L * env;
    mu += es0[i];
  }
  mu *= 0.125f;
  float var = 0.f;
  #pragma unroll
  for(int i = 0; i < 8; ++i){ float t = es0[i]-mu; var += t*t; }
  var *= 0.125f;
  float rs = rsqrtf(var + 1e-5f);
  float lnv[8];
  #pragma unroll
  for(int i = 0; i < 8; ++i) lnv[i] = (es0[i]-mu)*rs*P.lng[i] + P.lnb[i];
  // _ln_sv vector part (unit vector; normalize defensively)
  float un = sqrtf(fmaxf(ux*ux+uy*uy+uz*uz, EPSF));
  ux/=un; uy/=un; uz/=un;
  // We GVP (vi=1, h=1)
  float whs = P.wh[0];
  float vhx = ux*whs, vhy = uy*whs, vhz = uz*whs;
  float vn1 = sqrtf(fmaxf(vhx*vhx+vhy*vhy+vhz*vhz, EPSF));
  float ga = 0.f;
  for(int o = 0; o < 32; ++o){
    float acc = P.ws_b[o];
    #pragma unroll
    for(int c = 0; c < 8; ++c) acc += lnv[c]*P.ws_w[c*32 + o];
    acc += vn1 * P.ws_w[8*32 + o];
    P.es[(size_t)e*32 + o] = acc;
    ga += acc * P.wsv_w[o];
  }
  float gate = sigmoidf_(ga + P.wsv_b[0]);
  float wvs = P.wv[0];
  P.ev[(size_t)e*3+0] = vhx*wvs*gate;
  P.ev[(size_t)e*3+1] = vhy*wvs*gate;
  P.ev[(size_t)e*3+2] = vhz*wvs*gate;
  atomicAdd(&P.deg[d], 1.f);
}

// ---------------------------------------------------------------------------
struct NodeEmbP {
  const float* attrs; const float* lng; const float* lnb;
  const float* w; const float* b; float* hs;
};

__global__ __launch_bounds__(128) void k_node_embed(NodeEmbP P){
  int n = blockIdx.x, o = threadIdx.x;
  float a[5];
  #pragma unroll
  for(int c = 0; c < 5; ++c) a[c] = P.attrs[n*5 + c];
  float mu = (a[0]+a[1]+a[2]+a[3]+a[4]) * 0.2f;
  float var = 0.f;
  #pragma unroll
  for(int c = 0; c < 5; ++c){ float t = a[c]-mu; var += t*t; }
  var *= 0.2f;
  float rs = rsqrtf(var + 1e-5f);
  float acc = P.b[o];
  #pragma unroll
  for(int c = 0; c < 5; ++c)
    acc += ((a[c]-mu)*rs*P.lng[c] + P.lnb[c]) * P.w[c*128 + o];
  P.hs[(size_t)n*128 + o] = acc;
}

// ---------------------------------------------------------------------------
struct MsgP {
  const int* src; const int* dst;
  const float* hs; const float* hv;
  const float* es; const float* ev;
  const float* wh0; const _Float16* W0; const float* b0;   // msg0
  const float* wv0; const float* wsv0; const float* wsvb0;
  const float* wh1; const _Float16* W1; const float* b1;   // msg1
  const float* wv1; const float* wsv1; const float* wsvb1;
  float* aggs; float* aggv;
};

__global__ __launch_bounds__(64) void k_edge_msg(MsgP P){
  __shared__ _Float16 shA [2][16*352];
  __shared__ float    shMV[2][16*99];
  __shared__ float    shVH[2][16*99];
  __shared__ float    shS [2][16*128];
  const int wave = threadIdx.x >> 5;
  const int lane = threadIdx.x & 31;
  const int el = lane & 15;
  const int hf = lane >> 4;
  const int e  = (blockIdx.x*2 + wave)*16 + el;
  _Float16* A = shA[wave];
  float* MV = shMV[wave];
  float* VH = shVH[wave];
  float* S  = shS[wave];
  const int s_ = P.src[e], d_ = P.dst[e];

  // --- phase 1: stage mv, scalar features ---
  {
    int c0 = hf ? 17 : 0;
    int cn = hf ? 16 : 17;
    for(int ci = 0; ci < cn; ++ci){
      int c = c0 + ci;
      #pragma unroll
      for(int d3 = 0; d3 < 3; ++d3){
        float v;
        if(c < 16)       v = P.hv[(size_t)s_*48 + c*3 + d3];
        else if(c == 16) v = P.ev[(size_t)e*3 + d3];
        else             v = P.hv[(size_t)d_*48 + (c-17)*3 + d3];
        MV[el*99 + c*3 + d3] = v;
      }
    }
    for(int ks = hf*144; ks < hf*144 + 144; ++ks){
      float v;
      if(ks < 128)      v = P.hs[(size_t)s_*128 + ks];
      else if(ks < 160) v = P.es[(size_t)e*32 + (ks-128)];
      else              v = P.hs[(size_t)d_*128 + (ks-160)];
      A[el*352 + ks] = (_Float16)v;
    }
    int kp0 = hf ? 337 : 321;
    int kn  = hf ? 15 : 16;
    for(int i = 0; i < kn; ++i) A[el*352 + kp0 + i] = (_Float16)0.f;
  }
  __syncthreads();
  // --- vh = mv @ wh0 (33x33) + vector norms into A ---
  {
    int h0 = hf ? 17 : 0;
    int hn = hf ? 16 : 17;
    for(int hi = 0; hi < hn; ++hi){
      int h = h0 + hi;
      float a0=0.f, a1=0.f, a2=0.f;
      for(int c = 0; c < 33; ++c){
        float w = P.wh0[c*33 + h];
        a0 += MV[el*99 + c*3 + 0]*w;
        a1 += MV[el*99 + c*3 + 1]*w;
        a2 += MV[el*99 + c*3 + 2]*w;
      }
      VH[el*99 + h*3 + 0]=a0; VH[el*99 + h*3 + 1]=a1; VH[el*99 + h*3 + 2]=a2;
      A[el*352 + 288 + h] = (_Float16)sqrtf(fmaxf(a0*a0+a1*a1+a2*a2, EPSF));
    }
  }
  __syncthreads();
  // --- msg0 WMMA: [16x352] @ [352x128] ---
  for(int nt = 0; nt < 8; ++nt){
    v8f c = wmma_acc<11,8,352>(A, el, P.W0, nt);
    float bias = P.b0[nt*16 + el];
    #pragma unroll
    for(int v = 0; v < 8; ++v)
      S[(v + 8*hf)*128 + nt*16 + el] = c[v] + bias;
  }
  __syncthreads();
  // --- gate0, vo -> mv2; build A2 (silu(s), vn2) ---
  {
    const int o0 = hf*8;
    float g0[8];
    #pragma unroll
    for(int oo = 0; oo < 8; ++oo) g0[oo] = P.wsvb0[o0+oo];
    for(int j = 0; j < 128; ++j){
      float sj = S[el*128 + j];
      const float* wr = P.wsv0 + j*16 + o0;
      #pragma unroll
      for(int oo = 0; oo < 8; ++oo) g0[oo] += sj*wr[oo];
    }
    #pragma unroll
    for(int oo = 0; oo < 8; ++oo) g0[oo] = sigmoidf_(g0[oo]);
    float acc[8][3];
    #pragma unroll
    for(int oo = 0; oo < 8; ++oo){ acc[oo][0]=0.f; acc[oo][1]=0.f; acc[oo][2]=0.f; }
    for(int h = 0; h < 33; ++h){
      const float* wvr = P.wv0 + h*16 + o0;
      #pragma unroll
      for(int d3 = 0; d3 < 3; ++d3){
        float vv = VH[el*99 + h*3 + d3];
        #pragma unroll
        for(int oo = 0; oo < 8; ++oo) acc[oo][d3] += vv*wvr[oo];
      }
    }
    __syncthreads();
    #pragma unroll
    for(int oo = 0; oo < 8; ++oo)
      for(int d3 = 0; d3 < 3; ++d3)
        MV[el*99 + (o0+oo)*3 + d3] = acc[oo][d3]*g0[oo];
    for(int k = hf*64; k < hf*64 + 64; ++k)
      A[el*160 + k] = (_Float16)siluf_(S[el*128 + k]);
    for(int i = 0; i < 8; ++i) A[el*160 + 144 + hf*8 + i] = (_Float16)0.f;
  }
  __syncthreads();
  // --- vh2 = mv2 @ wh1 (16x16) ---
  {
    const int h20 = hf*8;
    float vacc[8][3];
    #pragma unroll
    for(int hi = 0; hi < 8; ++hi){
      int h2 = h20 + hi;
      float a0=0.f, a1=0.f, a2=0.f;
      for(int o = 0; o < 16; ++o){
        float w = P.wh1[o*16 + h2];
        a0 += MV[el*99 + o*3 + 0]*w;
        a1 += MV[el*99 + o*3 + 1]*w;
        a2 += MV[el*99 + o*3 + 2]*w;
      }
      vacc[hi][0]=a0; vacc[hi][1]=a1; vacc[hi][2]=a2;
      A[el*160 + 128 + h2] = (_Float16)sqrtf(fmaxf(a0*a0+a1*a1+a2*a2, EPSF));
    }
    __syncthreads();
    #pragma unroll
    for(int hi = 0; hi < 8; ++hi)
      for(int d3 = 0; d3 < 3; ++d3)
        VH[el*99 + (h20+hi)*3 + d3] = vacc[hi][d3];
  }
  __syncthreads();
  // --- msg1 WMMA: [16x160] @ [160x128] ---
  for(int nt = 0; nt < 8; ++nt){
    v8f c = wmma_acc<5,8,160>(A, el, P.W1, nt);
    float bias = P.b1[nt*16 + el];
    #pragma unroll
    for(int v = 0; v < 8; ++v)
      S[(v + 8*hf)*128 + nt*16 + el] = c[v] + bias;
  }
  __syncthreads();
  // --- gate1, vo2, scatter ---
  {
    const int o0 = hf*8;
    float g1[8];
    #pragma unroll
    for(int oo = 0; oo < 8; ++oo) g1[oo] = P.wsvb1[o0+oo];
    for(int j = 0; j < 128; ++j){
      float sj = S[el*128 + j];
      const float* wr = P.wsv1 + j*16 + o0;
      #pragma unroll
      for(int oo = 0; oo < 8; ++oo) g1[oo] += sj*wr[oo];
    }
    #pragma unroll
    for(int oo = 0; oo < 8; ++oo) g1[oo] = sigmoidf_(g1[oo]);
    float acc[8][3];
    #pragma unroll
    for(int oo = 0; oo < 8; ++oo){ acc[oo][0]=0.f; acc[oo][1]=0.f; acc[oo][2]=0.f; }
    for(int h2 = 0; h2 < 16; ++h2){
      const float* wr = P.wv1 + h2*16 + o0;
      #pragma unroll
      for(int d3 = 0; d3 < 3; ++d3){
        float vv = VH[el*99 + h2*3 + d3];
        #pragma unroll
        for(int oo = 0; oo < 8; ++oo) acc[oo][d3] += vv*wr[oo];
      }
    }
    float* as = P.aggs + (size_t)d_*128;
    for(int j = hf*64; j < hf*64 + 64; ++j) atomicAdd(&as[j], S[el*128 + j]);
    float* av = P.aggv + (size_t)d_*48;
    #pragma unroll
    for(int oo = 0; oo < 8; ++oo)
      for(int d3 = 0; d3 < 3; ++d3)
        atomicAdd(&av[(o0+oo)*3 + d3], acc[oo][d3]*g1[oo]);
  }
}

// ---------------------------------------------------------------------------
struct NodeP {
  const float* hs_in; const float* hv_in;
  const float* aggs; const float* aggv; const float* deg;
  const float* ln0g; const float* ln0b;
  const float* wh_f0; const _Float16* Wf0; const float* b_f0;
  const float* wv_f0; const float* wsv_f0; const float* wsvb_f0;
  const float* wh_f1; const _Float16* Wf1; const float* b_f1;
  const float* wv_f1; const float* wsv_f1; const float* wsvb_f1;
  const float* ln1g; const float* ln1b;
  float* hs_mid; float* hv_mid;
  float* hs_out; float* hv_out;
};

__global__ __launch_bounds__(32) void k_node_update(NodeP P){
  __shared__ float    sF[16*512];
  __shared__ _Float16 A [16*576];
  __shared__ float    V [16*96];
  __shared__ float    red[16*4];
  const int lane = threadIdx.x & 31;
  const int el = lane & 15, hf = lane >> 4;
  const int n = blockIdx.x*16 + el;
  const float dg = fmaxf(P.deg[n], 1.f);
  // --- residual + ln0 scalars ---
  {
    float p1 = 0.f, p2 = 0.f;
    for(int j = hf*64; j < hf*64 + 64; ++j){
      float v = P.hs_in[(size_t)n*128 + j] + P.aggs[(size_t)n*128 + j]/dg;
      p1 += v; p2 += v*v;
    }
    red[el*4 + hf*2 + 0] = p1;
    red[el*4 + hf*2 + 1] = p2;
  }
  __syncthreads();
  {
    float mu  = (red[el*4+0]+red[el*4+2]) * (1.f/128.f);
    float var = (red[el*4+1]+red[el*4+3]) * (1.f/128.f) - mu*mu;
    float rs  = rsqrtf(var + 1e-5f);
    for(int j = hf*64; j < hf*64 + 64; ++j){
      float v = P.hs_in[(size_t)n*128 + j] + P.aggs[(size_t)n*128 + j]/dg;
      float ln = (v-mu)*rs*P.ln0g[j] + P.ln0b[j];
      P.hs_mid[(size_t)n*128 + j] = ln;
      A[el*160 + j] = (_Float16)ln;
    }
  }
  __syncthreads();
  // --- residual + ln0 vectors ---
  {
    float vp = 0.f;
    for(int c = hf*8; c < hf*8 + 8; ++c){
      float s2 = 0.f;
      #pragma unroll
      for(int d3 = 0; d3 < 3; ++d3){
        float v = P.hv_in[(size_t)n*48 + c*3 + d3] + P.aggv[(size_t)n*48 + c*3 + d3]/dg;
        V[el*96 + c*3 + d3] = v;
        s2 += v*v;
      }
      vp += fmaxf(s2, EPSF);
    }
    red[el*4 + hf] = vp;
  }
  __syncthreads();
  {
    float inv = 1.f / sqrtf((red[el*4]+red[el*4+1]) * (1.f/16.f));
    for(int c = hf*8; c < hf*8 + 8; ++c)
      #pragma unroll
      for(int d3 = 0; d3 < 3; ++d3){
        float v = V[el*96 + c*3 + d3]*inv;
        V[el*96 + c*3 + d3] = v;
        P.hv_mid[(size_t)n*48 + c*3 + d3] = v;
      }
  }
  __syncthreads();
  // --- vh_ff0 = hv' @ wh_f0 (16x32) ---
  {
    float r[16][3];
    for(int hi = 0; hi < 16; ++hi){
      int h = hf*16 + hi;
      float a0=0.f, a1=0.f, a2=0.f;
      for(int c = 0; c < 16; ++c){
        float w = P.wh_f0[c*32 + h];
        a0 += V[el*96 + c*3 + 0]*w;
        a1 += V[el*96 + c*3 + 1]*w;
        a2 += V[el*96 + c*3 + 2]*w;
      }
      r[hi][0]=a0; r[hi][1]=a1; r[hi][2]=a2;
      A[el*160 + 128 + h] = (_Float16)sqrtf(fmaxf(a0*a0+a1*a1+a2*a2, EPSF));
    }
    __syncthreads();
    for(int hi = 0; hi < 16; ++hi){
      int h = hf*16 + hi;
      #pragma unroll
      for(int d3 = 0; d3 < 3; ++d3) V[el*96 + h*3 + d3] = r[hi][d3];
    }
  }
  __syncthreads();
  // --- ff0 WMMA: [16x160] @ [160x512] ---
  for(int nt = 0; nt < 32; ++nt){
    v8f c = wmma_acc<5,32,160>(A, el, P.Wf0, nt);
    float bias = P.b_f0[nt*16 + el];
    #pragma unroll
    for(int v = 0; v < 8; ++v)
      sF[(v + 8*hf)*512 + nt*16 + el] = c[v] + bias;
  }
  __syncthreads();
  // --- gate_ff0, mv_ff, build ff1 A ---
  {
    const int o0 = hf*16;
    float g[16];
    #pragma unroll
    for(int oo = 0; oo < 16; ++oo) g[oo] = P.wsvb_f0[o0+oo];
    for(int j = 0; j < 512; ++j){
      float sj = sF[el*512 + j];
      const float* wr = P.wsv_f0 + j*32 + o0;
      #pragma unroll
      for(int oo = 0; oo < 16; ++oo) g[oo] += sj*wr[oo];
    }
    #pragma unroll
    for(int oo = 0; oo < 16; ++oo) g[oo] = sigmoidf_(g[oo]);
    float mv[16][3];
    #pragma unroll
    for(int oo = 0; oo < 16; ++oo){ mv[oo][0]=0.f; mv[oo][1]=0.f; mv[oo][2]=0.f; }
    for(int h = 0; h < 32; ++h){
      const float* wr = P.wv_f0 + h*32 + o0;
      #pragma unroll
      for(int d3 = 0; d3 < 3; ++d3){
        float vv = V[el*96 + h*3 + d3];
        #pragma unroll
        for(int oo = 0; oo < 16; ++oo) mv[oo][d3] += vv*wr[oo];
      }
    }
    __syncthreads();
    #pragma unroll
    for(int oo = 0; oo < 16; ++oo)
      for(int d3 = 0; d3 < 3; ++d3)
        V[el*96 + (o0+oo)*3 + d3] = mv[oo][d3]*g[oo];
    for(int k = hf*256; k < hf*256 + 256; ++k)
      A[el*576 + k] = (_Float16)siluf_(sF[el*512 + k]);
    for(int i = 0; i < 16; ++i) A[el*576 + 544 + hf*16 + i] = (_Float16)0.f;
  }
  __syncthreads();
  // --- vh_ff1 = mv_ff @ wh_f1 (32x32) ---
  {
    float r[16][3];
    for(int hi = 0; hi < 16; ++hi){
      int h2 = hf*16 + hi;
      float a0=0.f, a1=0.f, a2=0.f;
      for(int o = 0; o < 32; ++o){
        float w = P.wh_f1[o*32 + h2];
        a0 += V[el*96 + o*3 + 0]*w;
        a1 += V[el*96 + o*3 + 1]*w;
        a2 += V[el*96 + o*3 + 2]*w;
      }
      r[hi][0]=a0; r[hi][1]=a1; r[hi][2]=a2;
      A[el*576 + 512 + h2] = (_Float16)sqrtf(fmaxf(a0*a0+a1*a1+a2*a2, EPSF));
    }
    __syncthreads();
    for(int hi = 0; hi < 16; ++hi)
      #pragma unroll
      for(int d3 = 0; d3 < 3; ++d3) V[el*96 + (hf*16+hi)*3 + d3] = r[hi][d3];
  }
  __syncthreads();
  // --- ff1 WMMA: [16x576] @ [576x128] ---
  for(int nt = 0; nt < 8; ++nt){
    v8f c = wmma_acc<18,8,576>(A, el, P.Wf1, nt);
    float bias = P.b_f1[nt*16 + el];
    #pragma unroll
    for(int v = 0; v < 8; ++v)
      sF[(v + 8*hf)*128 + nt*16 + el] = c[v] + bias;
  }
  __syncthreads();
  // --- gate_ff1, vo_ff1, residual + ln1, write out ---
  {
    const int o0 = hf*8;
    float g1[8];
    #pragma unroll
    for(int oo = 0; oo < 8; ++oo) g1[oo] = P.wsvb_f1[o0+oo];
    for(int j = 0; j < 128; ++j){
      float sj = sF[el*128 + j];
      const float* wr = P.wsv_f1 + j*16 + o0;
      #pragma unroll
      for(int oo = 0; oo < 8; ++oo) g1[oo] += sj*wr[oo];
    }
    #pragma unroll
    for(int oo = 0; oo < 8; ++oo) g1[oo] = sigmoidf_(g1[oo]);
    float vo[8][3];
    #pragma unroll
    for(int oo = 0; oo < 8; ++oo){ vo[oo][0]=0.f; vo[oo][1]=0.f; vo[oo][2]=0.f; }
    for(int h2 = 0; h2 < 32; ++h2){
      const float* wr = P.wv_f1 + h2*16 + o0;
      #pragma unroll
      for(int d3 = 0; d3 < 3; ++d3){
        float vv = V[el*96 + h2*3 + d3];
        #pragma unroll
        for(int oo = 0; oo < 8; ++oo) vo[oo][d3] += vv*wr[oo];
      }
    }
    float p1 = 0.f, p2 = 0.f;
    for(int j = hf*64; j < hf*64 + 64; ++j){
      float v = P.hs_mid[(size_t)n*128 + j] + sF[el*128 + j];
      p1 += v; p2 += v*v;
    }
    __syncthreads();
    red[el*4 + hf*2 + 0] = p1;
    red[el*4 + hf*2 + 1] = p2;
    __syncthreads();
    float mu1  = (red[el*4+0]+red[el*4+2]) * (1.f/128.f);
    float var1 = (red[el*4+1]+red[el*4+3]) * (1.f/128.f) - mu1*mu1;
    float rs1  = rsqrtf(var1 + 1e-5f);
    for(int j = hf*64; j < hf*64 + 64; ++j){
      float v = P.hs_mid[(size_t)n*128 + j] + sF[el*128 + j];
      P.hs_out[(size_t)n*128 + j] = (v-mu1)*rs1*P.ln1g[j] + P.ln1b[j];
    }
    float vpre[8][3]; float vp = 0.f;
    #pragma unroll
    for(int oo = 0; oo < 8; ++oo){
      float s2 = 0.f;
      #pragma unroll
      for(int d3 = 0; d3 < 3; ++d3){
        float v = P.hv_mid[(size_t)n*48 + (o0+oo)*3 + d3] + vo[oo][d3]*g1[oo];
        vpre[oo][d3] = v; s2 += v*v;
      }
      vp += fmaxf(s2, EPSF);
    }
    __syncthreads();
    red[el*4 + hf] = vp;
    __syncthreads();
    float inv1 = 1.f / sqrtf((red[el*4]+red[el*4+1]) * (1.f/16.f));
    #pragma unroll
    for(int oo = 0; oo < 8; ++oo)
      for(int d3 = 0; d3 < 3; ++d3)
        P.hv_out[(size_t)n*48 + (o0+oo)*3 + d3] = vpre[oo][d3]*inv1;
  }
}

// ---------------------------------------------------------------------------
struct OutP {
  const float* hs; const float* hv; const int* batch;
  const float* lng; const float* lnb;
  const float* wh; const float* ws_w; const float* ws_b;
  float* gacc; float* gcnt;
};

__global__ __launch_bounds__(64) void k_output(OutP P){
  __shared__ float s[128];
  __shared__ float hvn[48];
  __shared__ float vno[16];
  __shared__ float st[4];
  int n = blockIdx.x, t = threadIdx.x;
  s[t]      = P.hs[(size_t)n*128 + t];
  s[t + 64] = P.hs[(size_t)n*128 + 64 + t];
  if(t < 48) hvn[t] = P.hv[(size_t)n*48 + t];
  __syncthreads();
  if(t == 0){
    float p1 = 0.f, p2 = 0.f;
    for(int j = 0; j < 128; ++j){ p1 += s[j]; p2 += s[j]*s[j]; }
    float mu = p1*(1.f/128.f);
    float var = p2*(1.f/128.f) - mu*mu;
    st[0] = mu; st[1] = rsqrtf(var + 1e-5f);
    float vp = 0.f;
    for(int c = 0; c < 16; ++c){
      float s2 = 0.f;
      for(int d3 = 0; d3 < 3; ++d3){ float v = hvn[c*3+d3]; s2 += v*v; }
      vp += fmaxf(s2, EPSF);
    }
    st[2] = 1.f / sqrtf(vp*(1.f/16.f));
  }
  __syncthreads();
  float mu = st[0], rs = st[1], vinv = st[2];
  s[t]    = (s[t]   -mu)*rs*P.lng[t]    + P.lnb[t];
  s[t+64] = (s[t+64]-mu)*rs*P.lng[t+64] + P.lnb[t+64];
  if(t < 48) hvn[t] *= vinv;
  __syncthreads();
  if(t < 16){
    float a0=0.f, a1=0.f, a2=0.f;
    for(int c = 0; c < 16; ++c){
      float w = P.wh[c*16 + t];
      a0 += hvn[c*3+0]*w; a1 += hvn[c*3+1]*w; a2 += hvn[c*3+2]*w;
    }
    vno[t] = sqrtf(fmaxf(a0*a0+a1*a1+a2*a2, EPSF));
  }
  __syncthreads();
  float acc = P.ws_b[t];
  for(int k = 0; k < 128; ++k) acc += s[k]*P.ws_w[k*64 + t];
  for(int h = 0; h < 16; ++h)  acc += vno[h]*P.ws_w[(128+h)*64 + t];
  acc = siluf_(acc);
  int g = P.batch[n];
  atomicAdd(&P.gacc[g*64 + t], acc);
  if(t == 0) atomicAdd(&P.gcnt[g], 1.f);
}

__global__ __launch_bounds__(256) void k_div(const float* gacc, const float* gcnt, float* out){
  int i = blockIdx.x*256 + threadIdx.x;
  if(i < NGRAPH*NOUT){
    int g = i >> 6;
    out[i] = gacc[i] / fmaxf(gcnt[g], 1.f);
  }
}

// ---------------------------------------------------------------------------
extern "C" void kernel_launch(void* const* d_in, const int* in_sizes, int n_in,
                              void* d_out, int out_size, void* d_ws, size_t ws_size,
                              hipStream_t stream) {
  (void)in_sizes; (void)n_in; (void)out_size; (void)ws_size;
  const float* pos    = (const float*)d_in[0];
  const float* attrs  = (const float*)d_in[1];
  const float* shifts = (const float*)d_in[2];
  const int*   ei     = (const int*)d_in[3];
  const int*   batch  = (const int*)d_in[4];
  const int* src = ei;
  const int* dst = ei + NEDGE;
  // params: jax tree_leaves sorted-key order
  const float* We_wh   = (const float*)d_in[5];
  const float* We_wsb  = (const float*)d_in[6];
  const float* We_wsw  = (const float*)d_in[7];
  const float* We_wsvb = (const float*)d_in[8];
  const float* We_wsvw = (const float*)d_in[9];
  const float* We_wv   = (const float*)d_in[10];
  const float* Weln_b  = (const float*)d_in[11];
  const float* Weln_g  = (const float*)d_in[12];
  const float* Wo_wh   = (const float*)d_in[13];
  const float* Wo_wsb  = (const float*)d_in[14];
  const float* Wo_wsw  = (const float*)d_in[15];
  const float* Woln_b  = (const float*)d_in[16];
  const float* Woln_g  = (const float*)d_in[17];
  const float* Wv_wsb  = (const float*)d_in[18];
  const float* Wv_wsw  = (const float*)d_in[19];
  const float* Wvln_b  = (const float*)d_in[20];
  const float* Wvln_g  = (const float*)d_in[21];
  auto LP = [&](int L, int k){ return (const float*)d_in[22 + 28*L + k]; };
  // layer leaf offsets: ff0:0-5 ff1:6-11 ln0:12,13 ln1:14,15 msg0:16-21 msg1:22-27

  // workspace layout (f32 region then f16 region)
  float* W = (float*)d_ws;
  size_t off = 0;
  auto A_ = [&](size_t nel){ size_t o = off; off += (nel + 63) & ~(size_t)63; return o; };
  float* es   = W + A_((size_t)NEDGE*32);
  float* ev   = W + A_((size_t)NEDGE*3);
  float* deg  = W + A_(NNODE);
  float* hs0  = W + A_((size_t)NNODE*128);
  float* hs1  = W + A_((size_t)NNODE*128);
  float* hv0  = W + A_((size_t)NNODE*48);
  float* hv1  = W + A_((size_t)NNODE*48);
  float* hsm  = W + A_((size_t)NNODE*128);
  float* hvm  = W + A_((size_t)NNODE*48);
  float* aggs = W + A_((size_t)NNODE*128);
  float* aggv = W + A_((size_t)NNODE*48);
  float* gacc = W + A_(NGRAPH*NOUT);
  float* gcnt = W + A_(NGRAPH);
  _Float16* hb = (_Float16*)(W + off);
  size_t hoff = 0;
  auto H_ = [&](size_t nel){ size_t o = hoff; hoff += (nel + 63) & ~(size_t)63; return o; };
  _Float16* Wmsg0[2], *Wmsg1[2], *Wff0[2], *Wff1[2];
  for(int L = 0; L < 2; ++L){
    Wmsg0[L] = hb + H_(352*128);
    Wmsg1[L] = hb + H_(160*128);
    Wff0 [L] = hb + H_(160*512);
    Wff1 [L] = hb + H_(576*128);
  }

  auto zero = [&](float* p, size_t nel){
    k_zero<<<dim3((unsigned)((nel + 255)/256)), dim3(256), 0, stream>>>(p, (int)nel);
  };
  auto swz = [&](const float* srcw, _Float16* dstw, int K, int N, int KT){
    int total = KT*(N/16)*512;
    k_swizzle<<<dim3((total + 255)/256), dim3(256), 0, stream>>>(srcw, dstw, K, N, KT);
  };

  // weight prep (f16 swizzled B tiles)
  for(int L = 0; L < 2; ++L){
    swz(LP(L,18), Wmsg0[L], 321, 128, 11);   // msg0.ws_w (321x128) -> KP 352
    swz(LP(L,24), Wmsg1[L], 144, 128, 5);    // msg1.ws_w (144x128) -> KP 160
    swz(LP(L, 2), Wff0 [L], 160, 512, 5);    // ff0.ws_w  (160x512)
    swz(LP(L, 8), Wff1 [L], 544, 128, 18);   // ff1.ws_w  (544x128) -> KP 576
  }

  // init
  zero(deg, NNODE);
  zero(hv0, (size_t)NNODE*48);
  zero(gacc, NGRAPH*NOUT);
  zero(gcnt, NGRAPH);

  // edge features
  {
    EdgeFeatP P{pos, shifts, src, dst, Weln_g, Weln_b,
                We_wh, We_wsw, We_wsb, We_wv, We_wsvw, We_wsvb,
                es, ev, deg};
    k_edge_feat<<<dim3(NEDGE/256), dim3(256), 0, stream>>>(P);
  }
  // node embed
  {
    NodeEmbP P{attrs, Wvln_g, Wvln_b, Wv_wsw, Wv_wsb, hs0};
    k_node_embed<<<dim3(NNODE), dim3(128), 0, stream>>>(P);
  }

  float* hs_cur = hs0; float* hv_cur = hv0;
  float* hs_nxt = hs1; float* hv_nxt = hv1;
  for(int L = 0; L < 2; ++L){
    zero(aggs, (size_t)NNODE*128);
    zero(aggv, (size_t)NNODE*48);
    {
      MsgP P{src, dst, hs_cur, hv_cur, es, ev,
             LP(L,16), Wmsg0[L], LP(L,17), LP(L,21), LP(L,20), LP(L,19),
             LP(L,22), Wmsg1[L], LP(L,23), LP(L,27), LP(L,26), LP(L,25),
             aggs, aggv};
      k_edge_msg<<<dim3(NEDGE/32), dim3(64), 0, stream>>>(P);
    }
    {
      NodeP P{hs_cur, hv_cur, aggs, aggv, deg,
              LP(L,13), LP(L,12),
              LP(L,0), Wff0[L], LP(L,1), LP(L,5), LP(L,4), LP(L,3),
              LP(L,6), Wff1[L], LP(L,7), LP(L,11), LP(L,10), LP(L,9),
              LP(L,15), LP(L,14),
              hsm, hvm, hs_nxt, hv_nxt};
      k_node_update<<<dim3(NNODE/16), dim3(32), 0, stream>>>(P);
    }
    float* ts = hs_cur; hs_cur = hs_nxt; hs_nxt = ts;
    float* tv = hv_cur; hv_cur = hv_nxt; hv_nxt = tv;
  }

  // readout
  {
    OutP P{hs_cur, hv_cur, batch, Woln_g, Woln_b,
           Wo_wh, Wo_wsw, Wo_wsb, gacc, gcnt};
    k_output<<<dim3(NNODE), dim3(64), 0, stream>>>(P);
  }
  k_div<<<dim3((NGRAPH*NOUT + 255)/256), dim3(256), 0, stream>>>(gacc, gcnt, (float*)d_out);
}
